// PSMMix3DEquivEmbedding_88880053224004
// MI455X (gfx1250) — matrix operand order repaired and checked by reading, not compiled, not run on hardware
//
#include <hip/hip_runtime.h>
#include <math.h>

// Problem constants (match reference)
#define B_ 8
#define L_ 1024
#define D_ 1024
#define K_ 128
#define M_ (B_ * L_)                   // 8192 rows
#define SCALING_ 0.08838834764831845f  // 128^-0.5

typedef __attribute__((ext_vector_type(16))) _Float16 v16h;
typedef __attribute__((ext_vector_type(8)))  _Float16 h8;
typedef __attribute__((ext_vector_type(8)))  float    v8f;

union AFrag { v16h v; struct { h8 lo, hi; } p; };

// ---------------------------------------------------------------------------
// WMMA fragment loaders (contiguous k order used by this code, k = half*16 + s, wave32)
// A: 16x32 f16, row-major source [M, D_]:
//   lane = {m = lane&15, half = lane>>4}; elements 0..7 = K(half*16 + 0..7),
//   elements 8..15 = K(half*16 + 8 + 0..7)  -> two 16B contiguous loads.
// ---------------------------------------------------------------------------
__device__ inline v16h load_a_frag(const _Float16* A, int m0, int k0, int lane) {
    int m    = m0 + (lane & 15);
    int half = lane >> 4;
    const _Float16* p = A + (size_t)m * D_ + k0 + half * 16;
    AFrag f;
    f.p.lo = *(const h8*)(p);
    f.p.hi = *(const h8*)(p + 8);
    return f.v;
}

// B: 32x16 f16, source is column-major [N, D_] (pre-transposed weights):
//   lane = {n = lane&15, half = lane>>4}; elements j = K(half*16 + j)
//   -> one contiguous 32B load.
__device__ inline v16h load_b_frag(const _Float16* Bt, int n0, int k0, int lane) {
    int n    = n0 + (lane & 15);
    int half = lane >> 4;
    return *(const v16h*)(Bt + (size_t)n * D_ + k0 + half * 16);
}

// ---------------------------------------------------------------------------
// Weight convert + transpose: Wt[n*D + k] = (f16) W[k*D + n]
// ---------------------------------------------------------------------------
__global__ void k_convert_t(const float* __restrict__ W, _Float16* __restrict__ Wt) {
    int t = blockIdx.x * blockDim.x + threadIdx.x;  // D_*D_ threads
    int n = t / D_;
    int k = t - n * D_;
    Wt[(size_t)n * D_ + k] = (_Float16)W[(size_t)k * D_ + n];
}

// u[d] = sum_k pos_emb_w[k] * pos_feature_w[k*D + d]
__global__ void k_u(const float* __restrict__ pw, const float* __restrict__ pfw,
                    float* __restrict__ u) {
    int d = blockIdx.x * blockDim.x + threadIdx.x;
    float acc = 0.f;
    #pragma unroll 8
    for (int k = 0; k < K_; ++k) acc += pw[k] * pfw[(size_t)k * D_ + d];
    u[d] = acc;
}

// ---------------------------------------------------------------------------
// Per-row prep: mask, masked |pos|, x_base = embed + atom_emb, T = f16(time gather)
// One block per row (B*L blocks), 256 threads, 4 floats/thread.
// ---------------------------------------------------------------------------
__global__ void __launch_bounds__(256)
k_prep(const int* __restrict__ token, const int* __restrict__ node_attr,
       const unsigned char* __restrict__ is_mol, const float* __restrict__ pos,
       const int* __restrict__ tstep,
       const float* __restrict__ emb, const float* __restrict__ afeat,
       const float* __restrict__ ttab,
       float* __restrict__ out, float* __restrict__ mask_out,
       float* __restrict__ rnorm, _Float16* __restrict__ Tf) {
    int i  = blockIdx.x;     // row 0..M_-1
    int b  = i >> 10;        // i / L_
    int tid = threadIdx.x;
    int tok = token[i];
    bool pad = (tok == 0);

    if (tid == 0) {
        mask_out[i] = pad ? 1.0f : 0.0f;
        float x = pos[i * 3 + 0], y = pos[i * 3 + 1], z = pos[i * 3 + 2];
        rnorm[i] = pad ? 0.0f : sqrtf(x * x + y * y + z * z);
    }

    bool mol = (is_mol[b] != 0);
    int idx[8];
    #pragma unroll
    for (int f = 0; f < 8; ++f) idx[f] = node_attr[i * 9 + 1 + f];
    int ts = tstep[i];

    int d0 = tid * 4;
    float4 e = *(const float4*)(emb + (size_t)tok * D_ + d0);
    if (mol) {
        #pragma unroll
        for (int f = 0; f < 8; ++f) {
            float4 a = *(const float4*)(afeat + (size_t)idx[f] * D_ + d0);
            e.x += a.x; e.y += a.y; e.z += a.z; e.w += a.w;
        }
    }
    *(float4*)(out + (size_t)i * D_ + d0) = e;

    float4 t = *(const float4*)(ttab + (size_t)ts * D_ + d0);
    _Float16* tp = Tf + (size_t)i * D_ + d0;
    tp[0] = (_Float16)t.x; tp[1] = (_Float16)t.y;
    tp[2] = (_Float16)t.z; tp[3] = (_Float16)t.w;
}

// ---------------------------------------------------------------------------
// GEMM1: H = silu(T @ W1 + b1).  8 waves/WG, wave tile 16x64, K-step 32.
// Grid: (D_/64, M_/128), block 256.
// ---------------------------------------------------------------------------
__global__ void __launch_bounds__(256)
k_gemm_silu(const _Float16* __restrict__ A, const _Float16* __restrict__ Bt,
            const float* __restrict__ bias, _Float16* __restrict__ H) {
    int lane = threadIdx.x & 31;
    int wave = threadIdx.x >> 5;
    int m0 = blockIdx.y * 128 + wave * 16;
    int n0 = blockIdx.x * 64;

    v8f acc[4] = {v8f{}, v8f{}, v8f{}, v8f{}};
    for (int k0 = 0; k0 < D_; k0 += 32) {
        v16h a  = load_a_frag(A, m0, k0, lane);
        v16h b0 = load_b_frag(Bt, n0 +  0, k0, lane);
        v16h b1 = load_b_frag(Bt, n0 + 16, k0, lane);
        v16h b2 = load_b_frag(Bt, n0 + 32, k0, lane);
        v16h b3 = load_b_frag(Bt, n0 + 48, k0, lane);
        acc[0] = __builtin_amdgcn_wmma_f32_16x16x32_f16(false, a, false, b0, (short)0, acc[0], false, false);
        acc[1] = __builtin_amdgcn_wmma_f32_16x16x32_f16(false, a, false, b1, (short)0, acc[1], false, false);
        acc[2] = __builtin_amdgcn_wmma_f32_16x16x32_f16(false, a, false, b2, (short)0, acc[2], false, false);
        acc[3] = __builtin_amdgcn_wmma_f32_16x16x32_f16(false, a, false, b3, (short)0, acc[3], false, false);
    }

    int nl = lane & 15, half = lane >> 4;
    #pragma unroll
    for (int nn = 0; nn < 4; ++nn) {
        int n = n0 + nn * 16 + nl;
        float bv = bias[n];
        #pragma unroll
        for (int r = 0; r < 8; ++r) {
            int m = m0 + half * 8 + r;
            float v = acc[nn][r] + bv;
            float s = v / (1.0f + __expf(-v));  // silu
            H[(size_t)m * D_ + n] = (_Float16)s;
        }
    }
}

// GEMM2: out += H @ W2 + b2
__global__ void __launch_bounds__(256)
k_gemm_add(const _Float16* __restrict__ A, const _Float16* __restrict__ Bt,
           const float* __restrict__ bias, float* __restrict__ out) {
    int lane = threadIdx.x & 31;
    int wave = threadIdx.x >> 5;
    int m0 = blockIdx.y * 128 + wave * 16;
    int n0 = blockIdx.x * 64;

    v8f acc[4] = {v8f{}, v8f{}, v8f{}, v8f{}};
    for (int k0 = 0; k0 < D_; k0 += 32) {
        v16h a  = load_a_frag(A, m0, k0, lane);
        v16h b0 = load_b_frag(Bt, n0 +  0, k0, lane);
        v16h b1 = load_b_frag(Bt, n0 + 16, k0, lane);
        v16h b2 = load_b_frag(Bt, n0 + 32, k0, lane);
        v16h b3 = load_b_frag(Bt, n0 + 48, k0, lane);
        acc[0] = __builtin_amdgcn_wmma_f32_16x16x32_f16(false, a, false, b0, (short)0, acc[0], false, false);
        acc[1] = __builtin_amdgcn_wmma_f32_16x16x32_f16(false, a, false, b1, (short)0, acc[1], false, false);
        acc[2] = __builtin_amdgcn_wmma_f32_16x16x32_f16(false, a, false, b2, (short)0, acc[2], false, false);
        acc[3] = __builtin_amdgcn_wmma_f32_16x16x32_f16(false, a, false, b3, (short)0, acc[3], false, false);
    }

    int nl = lane & 15, half = lane >> 4;
    #pragma unroll
    for (int nn = 0; nn < 4; ++nn) {
        int n = n0 + nn * 16 + nl;
        float bv = bias[n];
        #pragma unroll
        for (int r = 0; r < 8; ++r) {
            size_t off = (size_t)(m0 + half * 8 + r) * D_ + n;
            out[off] += acc[nn][r] + bv;
        }
    }
}

// ---------------------------------------------------------------------------
// s[b,q] = sum_key softmax(dist*SCALING)[key] * rnorm_masked[key]
// Rank-1 collapse of attn @ pe. One block (256 thr) per query row.
// Max logit = SCALING (self-distance, dist=1), so shift is exact & stable.
// ---------------------------------------------------------------------------
__global__ void __launch_bounds__(256)
k_attn_s(const int* __restrict__ token, const float* __restrict__ pos,
         const float* __restrict__ rnorm, float* __restrict__ s) {
    int i = blockIdx.x;          // query row
    int b = i >> 10;
    int tid = threadIdx.x;
    bool qpad = (token[i] == 0);
    float qx = pos[i * 3 + 0], qy = pos[i * 3 + 1], qz = pos[i * 3 + 2];

    float num = 0.f, den = 0.f;
    if (!qpad) {
        for (int j = tid; j < L_; j += 256) {
            int kidx = b * L_ + j;
            if (token[kidx] != 0) {
                float dx = qx - pos[kidx * 3 + 0];
                float dy = qy - pos[kidx * 3 + 1];
                float dz = qz - pos[kidx * 3 + 2];
                float dist = sqrtf(dx * dx + dy * dy + dz * dz);
                float v = 1.0f / (dist + 1.0f);
                float w = __expf((v - 1.0f) * SCALING_);
                num += w * rnorm[kidx];
                den += w;
            }
        }
    }
    __shared__ float sn[256], sd[256];
    sn[tid] = num; sd[tid] = den;
    __syncthreads();
    for (int off = 128; off > 0; off >>= 1) {
        if (tid < off) { sn[tid] += sn[tid + off]; sd[tid] += sd[tid + off]; }
        __syncthreads();
    }
    if (tid == 0) s[i] = qpad ? 0.0f : sn[0] / sd[0];
}

// x[i,:] += s[i] * u[:]  (non-padding rows only)
__global__ void __launch_bounds__(256)
k_final(const int* __restrict__ token, const float* __restrict__ s,
        const float* __restrict__ u, float* __restrict__ out) {
    int i = blockIdx.x;
    if (token[i] == 0) return;
    int d0 = threadIdx.x * 4;
    float sv = s[i];
    float4 uv = *(const float4*)(u + d0);
    float4 o  = *(float4*)(out + (size_t)i * D_ + d0);
    o.x += sv * uv.x; o.y += sv * uv.y; o.z += sv * uv.z; o.w += sv * uv.w;
    *(float4*)(out + (size_t)i * D_ + d0) = o;
}

// ---------------------------------------------------------------------------
extern "C" void kernel_launch(void* const* d_in, const int* in_sizes, int n_in,
                              void* d_out, int out_size, void* d_ws, size_t ws_size,
                              hipStream_t stream) {
    (void)in_sizes; (void)n_in; (void)out_size; (void)ws_size;

    const int*           token  = (const int*)d_in[0];
    const int*           nattr  = (const int*)d_in[1];
    const unsigned char* is_mol = (const unsigned char*)d_in[2];
    const float*         pos    = (const float*)d_in[3];
    const int*           tstep  = (const int*)d_in[4];
    const float*         emb    = (const float*)d_in[5];
    const float*         afeat  = (const float*)d_in[6];
    const float*         ttab   = (const float*)d_in[7];
    const float*         w1     = (const float*)d_in[8];
    const float*         b1     = (const float*)d_in[9];
    const float*         w2     = (const float*)d_in[10];
    const float*         b2     = (const float*)d_in[11];
    const float*         pw     = (const float*)d_in[12];
    const float*         pfw    = (const float*)d_in[13];

    float* out = (float*)d_out;
    char*  ws  = (char*)d_ws;

    // Workspace layout (all offsets 256B-aligned)
    const size_t SZ_T   = (size_t)M_ * D_ * sizeof(_Float16);  // 16 MB
    const size_t SZ_W   = (size_t)D_ * D_ * sizeof(_Float16);  //  2 MB
    _Float16* Tf    = (_Float16*)(ws);
    _Float16* Hf    = (_Float16*)(ws + SZ_T);
    _Float16* W1t   = (_Float16*)(ws + 2 * SZ_T);
    _Float16* W2t   = (_Float16*)(ws + 2 * SZ_T + SZ_W);
    float*    u     = (float*)   (ws + 2 * SZ_T + 2 * SZ_W);
    float*    rnorm = u + D_;
    float*    sbuf  = rnorm + M_;
    float*    mask_out = out + (size_t)M_ * D_;

    // 1. weight convert/transpose + rank-1 u precompute
    k_convert_t<<<dim3((D_ * D_) / 256), dim3(256), 0, stream>>>(w1, W1t);
    k_convert_t<<<dim3((D_ * D_) / 256), dim3(256), 0, stream>>>(w2, W2t);
    k_u<<<dim3(D_ / 256), dim3(256), 0, stream>>>(pw, pfw, u);

    // 2. gathers + mask + rnorm + f16 time rows
    k_prep<<<dim3(M_), dim3(256), 0, stream>>>(token, nattr, is_mol, pos, tstep,
                                               emb, afeat, ttab,
                                               out, mask_out, rnorm, Tf);

    // 3. time MLP on WMMA
    dim3 gg(D_ / 64, M_ / 128);
    k_gemm_silu<<<gg, dim3(256), 0, stream>>>(Tf, W1t, b1, Hf);
    k_gemm_add <<<gg, dim3(256), 0, stream>>>(Hf, W2t, b2, out);

    // 4. collapsed positional attention + rank-1 add
    k_attn_s<<<dim3(M_), dim3(256), 0, stream>>>(token, pos, rnorm, sbuf);
    k_final <<<dim3(M_), dim3(256), 0, stream>>>(token, sbuf, u, out);
}
